// Detection_Network_69922067579288
// MI455X (gfx1250) — matrix operand order repaired and checked by reading, not compiled, not run on hardware
//
#include <hip/hip_runtime.h>
#include <hip/hip_bf16.h>
#include <cstdint>

typedef __bf16          v16bf   __attribute__((ext_vector_type(16)));
typedef float           v8f     __attribute__((ext_vector_type(8)));
typedef unsigned short  v16us   __attribute__((ext_vector_type(16)));
typedef unsigned short  us8     __attribute__((ext_vector_type(8)));

// ---------------------------------------------------------------------------
// fp32 -> bf16, round-to-nearest-even
// ---------------------------------------------------------------------------
__device__ __forceinline__ unsigned short f2bf(float f) {
    unsigned int u = __float_as_uint(f);
    u += 0x7FFFu + ((u >> 16) & 1u);
    return (unsigned short)(u >> 16);
}

// generic (flat) LDS pointer -> 32-bit LDS byte offset (addr[31:0] per aperture rules)
__device__ __forceinline__ unsigned lds_off(const void* p) {
    return (unsigned)(uintptr_t)p;
}

// ---------------------------------------------------------------------------
// Kernel 0: build integer rois (256 x 5) and mirror them into d_out tail
// ---------------------------------------------------------------------------
__global__ void make_rois_kernel(const float* __restrict__ sr,
                                 int* __restrict__ rois_ws,
                                 int* __restrict__ rois_out) {
    int t = threadIdx.x;
    if (t < 256) {
        const float S = 0.0625f;
        float a0 = sr[t * 4 + 0];
        float a1 = sr[t * 4 + 1];
        float a2 = sr[t * 4 + 2];
        float a3 = sr[t * 4 + 3];
        int r1 = (int)(a1 * S);
        int r2 = (int)(a0 * S);
        int r3 = (int)(a3 * S);
        int r4 = (int)(a2 * S);
        rois_ws[t * 5 + 0] = 0;  rois_out[t * 5 + 0] = 0;
        rois_ws[t * 5 + 1] = r1; rois_out[t * 5 + 1] = r1;
        rois_ws[t * 5 + 2] = r2; rois_out[t * 5 + 2] = r2;
        rois_ws[t * 5 + 3] = r3; rois_out[t * 5 + 3] = r3;
        rois_ws[t * 5 + 4] = r4; rois_out[t * 5 + 4] = r4;
    }
}

// ---------------------------------------------------------------------------
// Kernel 1: ROI max-pool (1,512,50,50) -> feat (256 x 25088) stored as bf16
// (the GEMM rounds A to bf16 anyway; storing bf16 halves A traffic and makes
//  the GEMM A-stage a pure copy -> async-LDS eligible)
// ---------------------------------------------------------------------------
__global__ void roipool_kernel(const float* __restrict__ fm,
                               const int* __restrict__ rois,
                               unsigned short* __restrict__ feat) {
    __shared__ int s_rs[7], s_re[7], s_cs[7], s_ce[7];
    int r = blockIdx.x;
    const int* roi = rois + r * 5;
    int x1 = roi[1], y1 = roi[2], x2 = roi[3], y2 = roi[4];
    int h = y2 - y1 + 1;
    int w = x2 - x1 + 1;
    int t = threadIdx.x;
    if (t < 7) {
        s_rs[t] = y1 + (t * h) / 7;
        s_re[t] = y1 + ((t + 1) * h + 6) / 7 - 1;
        s_cs[t] = x1 + (t * w) / 7;
        s_ce[t] = x1 + ((t + 1) * w + 6) / 7 - 1;
    }
    __syncthreads();
    for (int o = t; o < 25088; o += blockDim.x) {
        int c   = o / 49;
        int rem = o % 49;
        int jr  = rem / 7;
        int jc  = rem % 7;
        int rs = s_rs[jr], re = s_re[jr];
        int cs = s_cs[jc], ce = s_ce[jc];
        float m = -3.402823466e38f;  // finfo(f32).min == empty-bin value
        const float* p = fm + c * 2500;
        for (int y = rs; y <= re; ++y) {
            const float* row = p + y * 50;
            for (int x = cs; x <= ce; ++x)
                m = fmaxf(m, row[x]);
        }
        feat[(size_t)r * 25088 + o] = f2bf(m);
    }
}

// ---------------------------------------------------------------------------
// Kernel 2: C[256][N] = A(bf16)[256][K] * B(f32)[K][N] + bias.
// Block tile 256(M) x 64(N); 512 threads = 16 wave32 (4x4); wave tile 64x16.
// Double-buffered LDS; A tile copied with global_load_async_to_lds_b128
// (ASYNCcnt), B tile converted f32->bf16 and stored transposed.
// Fragment gathers are contiguous ds_load_b128 per the ISA VGPR layouts.
// ---------------------------------------------------------------------------
template <bool OUT_BF16>
__launch_bounds__(512)
__global__ void gemm_bf16_kernel(const unsigned short* __restrict__ A,
                                 const float* __restrict__ B,
                                 const float* __restrict__ bias,
                                 void* __restrict__ Cout,
                                 int N, int K, int lda) {
    __shared__ unsigned short sA[2][256 * 40];  // 2 x 20 KB (32 K, pad->40)
    __shared__ unsigned short sB[2][64 * 40];   // 2 x  5 KB (col-major)

    const int t    = threadIdx.x;
    const int lane = t & 31;
    const int wid  = t >> 5;
    const int hi   = lane >> 4;
    const int lm   = lane & 15;
    const int wm   = wid >> 2;       // rows wm*64
    const int wn   = wid & 3;        // cols wn*16
    const int nbase = blockIdx.x * 64;

    // A async staging: 1024 x 16B units; 2 per thread
    const int rowA0 = t >> 2;             // 0..127
    const int segA0 = (t & 3) * 8;        // bf16 elems within 32-K chunk
    const int rowA1 = rowA0 + 128;
    // B staging: one float4 per thread
    const int kB  = t >> 4;               // 0..31
    const int n4B = (t & 15) * 4;

    v8f acc[4] = {};

    auto asyncA = [&](int k0, int buf) {
        unsigned off0 = (unsigned)(((size_t)rowA0 * lda + k0 + segA0) * 2);
        unsigned off1 = (unsigned)(((size_t)rowA1 * lda + k0 + segA0) * 2);
        unsigned l0 = lds_off(&sA[buf][rowA0 * 40 + segA0]);
        unsigned l1 = lds_off(&sA[buf][rowA1 * 40 + segA0]);
        asm volatile("global_load_async_to_lds_b128 %0, %1, %2 offset:0"
                     :: "v"(l0), "v"(off0), "s"(A) : "memory");
        asm volatile("global_load_async_to_lds_b128 %0, %1, %2 offset:0"
                     :: "v"(l1), "v"(off1), "s"(A) : "memory");
    };
    auto loadB = [&](int k0) -> float4 {
        return *(const float4*)(B + (size_t)(k0 + kB) * N + nbase + n4B);
    };
    auto storeB = [&](float4 v, int buf) {
        sB[buf][(n4B + 0) * 40 + kB] = f2bf(v.x);
        sB[buf][(n4B + 1) * 40 + kB] = f2bf(v.y);
        sB[buf][(n4B + 2) * 40 + kB] = f2bf(v.z);
        sB[buf][(n4B + 3) * 40 + kB] = f2bf(v.w);
    };
    auto compute = [&](int buf) {
        // B fragment: column wn*16+lm, K = 8hi..+7 then 16+8hi..+7
        v16bf bfrag;
        {
            int n = wn * 16 + lm;
            us8 lo = *(const us8*)&sB[buf][n * 40 + 8 * hi];
            us8 h8 = *(const us8*)&sB[buf][n * 40 + 16 + 8 * hi];
            v16us u;
#pragma unroll
            for (int i = 0; i < 8; ++i) { u[i] = lo[i]; u[i + 8] = h8[i]; }
            bfrag = __builtin_bit_cast(v16bf, u);
        }
#pragma unroll
        for (int mt = 0; mt < 4; ++mt) {
            int row = wm * 64 + mt * 16 + lm;
            us8 lo = *(const us8*)&sA[buf][row * 40 + 8 * hi];       // K=8hi..+7
            us8 h8 = *(const us8*)&sA[buf][row * 40 + 16 + 8 * hi];  // K=16+8hi..+7
            v16us u;
#pragma unroll
            for (int i = 0; i < 8; ++i) { u[i] = lo[i]; u[i + 8] = h8[i]; }
            v16bf afrag = __builtin_bit_cast(v16bf, u);
            acc[mt] = __builtin_amdgcn_wmma_f32_16x16x32_bf16(
                false, afrag, false, bfrag, (short)0, acc[mt], false, false);
        }
    };

    // ---- prologue: stage chunk 0 into buffer 0
    {
        float4 vb = loadB(0);
        asyncA(0, 0);
        storeB(vb, 0);
    }
    asm volatile("s_wait_asynccnt 0x0" ::: "memory");
    __syncthreads();

    const int nk = K / 32;
    for (int kc = 0; kc < nk; ++kc) {
        const int cur = kc & 1;
        const bool more = (kc + 1 < nk);
        float4 vbn;
        if (more) {
            asyncA((kc + 1) * 32, cur ^ 1);   // DMA overlaps the WMMAs below
            vbn = loadB((kc + 1) * 32);       // W stream in flight during math
        }
        compute(cur);
        if (more) storeB(vbn, cur ^ 1);
        asm volatile("s_wait_asynccnt 0x0" ::: "memory");
        __syncthreads();
    }

    // ---- epilogue: C layout -> VGPR v holds (M = v + 8*hi, N = lm)
#pragma unroll
    for (int mt = 0; mt < 4; ++mt) {
        int col = nbase + wn * 16 + lm;
        float bv = bias[col];
#pragma unroll
        for (int v = 0; v < 8; ++v) {
            int row = wm * 64 + mt * 16 + v + 8 * hi;
            float r = acc[mt][v] + bv;
            if (OUT_BF16)
                ((unsigned short*)Cout)[(size_t)row * N + col] = f2bf(r);
            else
                ((float*)Cout)[(size_t)row * N + col] = r;
        }
    }
}

// ---------------------------------------------------------------------------
// Kernel 3: tiny heads  loc = k @ W_loc + b_loc (44), sc = k @ W_sc + b_sc (11)
// ---------------------------------------------------------------------------
__global__ void heads_kernel(const float* __restrict__ K2,
                             const float* __restrict__ Wl,
                             const float* __restrict__ bl,
                             const float* __restrict__ Wsc,
                             const float* __restrict__ bsc,
                             float* __restrict__ out) {
    int r = blockIdx.x;
    int t = threadIdx.x;
    const float* a = K2 + (size_t)r * 4096;
    if (t < 44) {
        float acc = bl[t];
        for (int k = 0; k < 4096; ++k)
            acc = fmaf(a[k], Wl[(size_t)k * 44 + t], acc);
        out[r * 44 + t] = acc;
    } else if (t < 55) {
        int c = t - 44;
        float acc = bsc[c];
        for (int k = 0; k < 4096; ++k)
            acc = fmaf(a[k], Wsc[(size_t)k * 11 + c], acc);
        out[256 * 44 + r * 11 + c] = acc;
    }
}

// ---------------------------------------------------------------------------
// Launch
// ---------------------------------------------------------------------------
extern "C" void kernel_launch(void* const* d_in, const int* in_sizes, int n_in,
                              void* d_out, int out_size, void* d_ws, size_t ws_size,
                              hipStream_t stream) {
    const float* sample_roi = (const float*)d_in[0];
    const float* out_map    = (const float*)d_in[1];
    const float* W1         = (const float*)d_in[2];
    const float* b1         = (const float*)d_in[3];
    const float* W2         = (const float*)d_in[4];
    const float* b2         = (const float*)d_in[5];
    const float* W_loc      = (const float*)d_in[6];
    const float* b_loc      = (const float*)d_in[7];
    const float* W_sc       = (const float*)d_in[8];
    const float* b_sc       = (const float*)d_in[9];

    char* ws = (char*)d_ws;
    const size_t FEAT_BYTES = (size_t)256 * 25088 * 2;   // bf16: 12,845,056
    const size_t K1_BYTES   = (size_t)256 * 4096 * 2;    // bf16:  2,097,152
    const size_t K2_BYTES   = (size_t)256 * 4096 * 4;    // f32:   4,194,304
    unsigned short* feat = (unsigned short*)(ws);
    unsigned short* k1   = (unsigned short*)(ws + FEAT_BYTES);
    float*          k2   = (float*)(ws + FEAT_BYTES + K1_BYTES);
    int*            rois = (int*)(ws + FEAT_BYTES + K1_BYTES + K2_BYTES);

    float* out      = (float*)d_out;
    int*   rois_out = (int*)d_out + 256 * 55;            // after loc+score

    make_rois_kernel<<<1, 256, 0, stream>>>(sample_roi, rois, rois_out);
    roipool_kernel<<<256, 256, 0, stream>>>(out_map, rois, feat);
    // k1 = feat @ W1 + b1 : K=25088, W1 (411 MB) streamed from HBM exactly once
    gemm_bf16_kernel<true ><<<64, 512, 0, stream>>>(feat, W1, b1, (void*)k1, 4096, 25088, 25088);
    // k2 = k1 @ W2 + b2 : K=4096
    gemm_bf16_kernel<false><<<64, 512, 0, stream>>>(k1,   W2, b2, (void*)k2, 4096, 4096, 4096);
    heads_kernel<<<256, 64, 0, stream>>>(k2, W_loc, b_loc, W_sc, b_sc, out);
}
